// DescriptionEmbedding_61005715472615
// MI455X (gfx1250) — matrix operand order repaired and checked by reading, not compiled, not run on hardware
//
#include <hip/hip_runtime.h>
#include <hip/hip_bf16.h>
#include <math.h>

// ---------------------------------------------------------------------------
// Problem constants (from the reference)
// ---------------------------------------------------------------------------
#define NFEAT   50000
#define NHID    50000
#define DIM     128
#define ADIM    128
#define KCONN   20
#define BATCH   256
#define KTOT    50000          // K of final GEMM (= NFEAT)
#define NCHUNK  25
#define KCHUNK  2000           // 25 * 2000 = 50000
#define MTILES_H 5             // M-tiles per block in gemm_h (3125 = 5 * 625)

typedef __attribute__((ext_vector_type(16))) __bf16 v16bf;
typedef __attribute__((ext_vector_type(8)))  float  v8f;

// ---------------------------------------------------------------------------
// Kernel 1: H[row, a] = sum_d A[row, d] * W[wOff + d, a]
// The 128x128 weight slice is converted to bf16 and staged in LDS once per
// block; each block computes 5 consecutive 16-row M-tiles (8 waves cover the
// 128-wide N), so the staging is amortized over 40 WMMA tiles.
// ---------------------------------------------------------------------------
__global__ __launch_bounds__(256)
void gemm_h_kernel(const float* __restrict__ Arows,   // [nrows, 128] f32
                   const float* __restrict__ W,       // w_kernel [256, 128] f32
                   int wOff,                          // 0 (top half) or 128 (bottom)
                   float* __restrict__ Hout)          // [nrows, 128] f32
{
    __shared__ __bf16 sW[DIM * ADIM];                 // 32 KB, layout [k][n]

    const int t = threadIdx.x;

    // ---- stage weight slice (rows wOff..wOff+127 are contiguous) ----
    {
        const float* src = W + (size_t)wOff * ADIM + t * 64;
        __bf16*      dst = &sW[t * 64];
        #pragma unroll
        for (int j = 0; j < 64; ++j)
            dst[j] = (__bf16)src[j];
    }
    __syncthreads();

    const int wave = t >> 5;
    const int lane = t & 31;
    const int hi   = lane >> 4;            // half-wave select
    const int l16  = lane & 15;
    const int n    = wave * 16 + l16;

    #pragma unroll
    for (int mt = 0; mt < MTILES_H; ++mt) {
        const int rowTile = (blockIdx.x * MTILES_H + mt) * 16;
        const int row     = rowTile + l16;

        v8f acc = {};
        #pragma unroll
        for (int kk = 0; kk < DIM; kk += 32) {
            v16bf a, b;
            // A fragment: lane (row) holds K = kk + {0..7,16..23} + 8*hi
            const float* Ap = Arows + (size_t)row * DIM + kk + 8 * hi;
            #pragma unroll
            for (int i = 0; i < 8; ++i) {
                a[i]     = (__bf16)Ap[i];
                a[8 + i] = (__bf16)Ap[16 + i];
            }
            // B fragment from LDS: lane (col n) holds K = kk + {0..7,16..23} + 8*hi
            #pragma unroll
            for (int i = 0; i < 16; ++i)
                b[i] = sW[(kk + 8 * hi + (i & 7) + ((i >> 3) << 4)) * ADIM + n];

            acc = __builtin_amdgcn_wmma_f32_16x16x32_bf16(
                      false, a, false, b, (short)0, acc, false, false);
        }
        // C/D layout: VGPR v, lane l -> row = v + 8*hi, col = l16
        float* Op = Hout + (size_t)(rowTile + 8 * hi) * ADIM + n;
        #pragma unroll
        for (int v = 0; v < 8; ++v)
            Op[(size_t)v * ADIM] = acc[v];
    }
}

// ---------------------------------------------------------------------------
// Kernel 2: per-feature attention scores + segment softmax + context row.
// One 128-thread block (4 waves) per feature; thread a owns attention dim a.
//   s_k = sum_a u[a] * tanh(h_top[i,a] + bias[a] + h_all[e_k, a])
//   w_k = softmax_k(s_k);  context[i,a] = sum_k w_k * all_emb[e_k, a]
// ---------------------------------------------------------------------------
__global__ __launch_bounds__(128)
void attn_kernel(const float* __restrict__ h_top,     // [N, 128]
                 const float* __restrict__ h_all,     // [N+H, 128]
                 const float* __restrict__ feat_emb,  // [N, 128]
                 const float* __restrict__ hidden_emb,// [H, 128]
                 const float* __restrict__ w_bias,    // [128]
                 const float* __restrict__ u_kernel,  // [128]
                 const int*   __restrict__ conn_idx,  // [N*K]
                 __bf16*      __restrict__ context)   // [N, 128] bf16
{
    const int i = blockIdx.x;
    const int a = threadIdx.x;

    __shared__ float sPart[4];
    __shared__ float sS[KCONN];
    __shared__ int   sE[KCONN];

    if (a < KCONN)
        sE[a] = conn_idx[(size_t)i * KCONN + a];

    const float ht = h_top[(size_t)i * ADIM + a] + w_bias[a];
    const float u  = u_kernel[a];
    __syncthreads();

    for (int k = 0; k < KCONN; ++k) {
        const int e = sE[k];
        if (k + 1 < KCONN)   // hide the L2 gather latency of the next row
            __builtin_prefetch(&h_all[(size_t)sE[k + 1] * ADIM + a], 0, 1);
        float v = tanhf(ht + h_all[(size_t)e * ADIM + a]) * u;
        // wave32 reduce, then 4-wave LDS reduce
        #pragma unroll
        for (int off = 16; off > 0; off >>= 1)
            v += __shfl_down(v, off, 32);
        if ((a & 31) == 0) sPart[a >> 5] = v;
        __syncthreads();
        if (a == 0) sS[k] = sPart[0] + sPart[1] + sPart[2] + sPart[3];
        __syncthreads();
    }

    // stable softmax over K=20, computed redundantly per thread
    float mx = sS[0];
    #pragma unroll
    for (int k = 1; k < KCONN; ++k) mx = fmaxf(mx, sS[k]);
    float wk[KCONN], wsum = 0.0f;
    #pragma unroll
    for (int k = 0; k < KCONN; ++k) { wk[k] = expf(sS[k] - mx); wsum += wk[k]; }
    const float inv = 1.0f / wsum;

    // weighted aggregation of neighbor embeddings (gathers are L2-resident)
    float acc = 0.0f;
    for (int k = 0; k < KCONN; ++k) {
        const int e = sE[k];
        const float* src = (e < NFEAT)
                         ? (feat_emb   + (size_t)e           * DIM)
                         : (hidden_emb + (size_t)(e - NFEAT) * DIM);
        acc += (wk[k] * inv) * src[a];
    }
    context[(size_t)i * DIM + a] = (__bf16)acc;
}

// ---------------------------------------------------------------------------
// Kernel 3: split-K WMMA GEMM  part[c] = values_tile @ context_tile
// out[256,128] = values[256,50000] @ context[50000,128].
// Per 32-K step the 32x128 bf16 context tile (8 KB) is staged into LDS with
// CDNA5 async copies (GLOBAL_LOAD_ASYNC_TO_LDS_B128, drained by
// s_wait_asynccnt); B fragments then come from LDS. Chunk tails
// (2000 % 32 != 0) are zero-masked at fragment build; the <=8KB staging
// overread past the chunk stays inside the workspace.
// ---------------------------------------------------------------------------
__global__ __launch_bounds__(256)
void gemm_out_kernel(const float*  __restrict__ values,   // [256, 50000] f32
                     const __bf16* __restrict__ context,  // [50000, 128] bf16
                     float*        __restrict__ part)     // [NCHUNK, 256, 128]
{
    __shared__ __bf16 sB[32 * DIM];      // 8 KB: context[kk..kk+31, 0..127]

    const int t    = threadIdx.x;
    const int wave = t >> 5;
    const int lane = t & 31;
    const int hi   = lane >> 4;
    const int l16  = lane & 15;

    const int mTile = blockIdx.x & 15;       // 16 M-tiles
    const int chunk = blockIdx.x >> 4;       // 25 K-chunks
    const int m0 = mTile * 16;
    const int kBeg = chunk * KCHUNK;
    const int kEnd = kBeg + KCHUNK;          // always <= KTOT

    const int row = m0 + l16;
    const int n   = wave * 16 + l16;
    const __bf16 zero = (__bf16)0.0f;

    // LDS byte offset for this thread's 32-byte staging slice
    // (low 32 bits of the generic LDS pointer are the wave-relative address)
    const unsigned ldsOff = (unsigned)(size_t)(&sB[0]) + (unsigned)(t * 32);

    v8f acc = {};
    for (int kk = kBeg; kk < kEnd; kk += 32) {
        // ---- async-stage the 8 KB context tile: thread t copies 32 bytes ----
        const unsigned long long gaddr =
            (unsigned long long)(size_t)context
            + ((unsigned long long)kk * DIM) * 2ull
            + (unsigned long long)(t * 32);
        asm volatile("global_load_async_to_lds_b128 %0, %1, off"
                     :: "v"(ldsOff), "v"(gaddr) : "memory");
        asm volatile("global_load_async_to_lds_b128 %0, %1, off offset:16"
                     :: "v"(ldsOff), "v"(gaddr) : "memory");
        asm volatile("s_wait_asynccnt 0x0" ::: "memory");
        __syncthreads();                       // all waves' tiles landed

        v16bf a, b;
        // B fragment from LDS (mask chunk tail)
        #pragma unroll
        for (int i = 0; i < 16; ++i) {
            const int kb = kk + 8 * hi + (i & 7) + ((i >> 3) << 4);
            const __bf16 bv = sB[(8 * hi + (i & 7) + ((i >> 3) << 4)) * DIM + n];
            b[i] = (kb < kEnd) ? bv : zero;
        }
        __syncthreads();                       // reads done before next staging

        // A fragment direct from global with in-register f32->bf16
        #pragma unroll
        for (int i = 0; i < 8; ++i) {
            const int k0 = kk + 8 * hi + i;
            const int k1 = k0 + 16;
            const float a0 = values[(size_t)row * KTOT + (k0 < kEnd ? k0 : kBeg)];
            const float a1 = values[(size_t)row * KTOT + (k1 < kEnd ? k1 : kBeg)];
            a[i]     = (k0 < kEnd) ? (__bf16)a0 : zero;
            a[8 + i] = (k1 < kEnd) ? (__bf16)a1 : zero;
        }

        acc = __builtin_amdgcn_wmma_f32_16x16x32_bf16(
                  false, a, false, b, (short)0, acc, false, false);
    }

    float* Op = part + ((size_t)chunk * BATCH + (m0 + 8 * hi)) * DIM + n;
    #pragma unroll
    for (int v = 0; v < 8; ++v)
        Op[(size_t)v * DIM] = acc[v];
}

// ---------------------------------------------------------------------------
// Kernel 4: deterministic chunk reduction -> d_out
// ---------------------------------------------------------------------------
__global__ __launch_bounds__(256)
void reduce_kernel(const float* __restrict__ part, float* __restrict__ out)
{
    const int idx = blockIdx.x * 256 + threadIdx.x;   // 0 .. 256*128-1
    float s = 0.0f;
    #pragma unroll
    for (int c = 0; c < NCHUNK; ++c)
        s += part[(size_t)c * (BATCH * DIM) + idx];
    out[idx] = s;
}

// ---------------------------------------------------------------------------
// Launch
// ---------------------------------------------------------------------------
extern "C" void kernel_launch(void* const* d_in, const int* in_sizes, int n_in,
                              void* d_out, int out_size, void* d_ws, size_t ws_size,
                              hipStream_t stream) {
    (void)in_sizes; (void)n_in; (void)out_size; (void)ws_size;

    const float* values     = (const float*)d_in[0];   // [256, 50000]
    const float* feat_emb   = (const float*)d_in[1];   // [50000, 128]
    const float* hidden_emb = (const float*)d_in[2];   // [50000, 128]
    const float* w_kernel   = (const float*)d_in[3];   // [256, 128]
    const float* w_bias     = (const float*)d_in[4];   // [128]
    const float* u_kernel   = (const float*)d_in[5];   // [128, 1]
    const int*   conn_idx   = (const int*)d_in[6];     // [1000000]
    // d_in[7] (partition) is implicit: connection j belongs to feature j/20.

    float* out = (float*)d_out;                        // [256, 128]

    // Workspace layout (all offsets 256B-aligned):
    //   h_top   : [N,   128] f32   = 25,600,000 B
    //   h_all   : [N+H, 128] f32   = 51,200,000 B
    //   context : [N,   128] bf16  = 12,800,000 B
    //   part    : [25, 256, 128] f32 = 3,276,800 B
    char* ws = (char*)d_ws;
    float*  h_top   = (float*)(ws);
    float*  h_all   = (float*)(ws + 25600000ull);
    __bf16* context = (__bf16*)(ws + 25600000ull + 51200000ull);
    float*  part    = (float*)(ws + 25600000ull + 51200000ull + 12800000ull);

    // Stage 1: h_top = feat_emb @ W[0:128];  h_all = concat(feat,hidden) @ W[128:256]
    gemm_h_kernel<<<NFEAT / (16 * MTILES_H), 256, 0, stream>>>(feat_emb, w_kernel, 0,   h_top);
    gemm_h_kernel<<<NFEAT / (16 * MTILES_H), 256, 0, stream>>>(feat_emb, w_kernel, 128, h_all);
    gemm_h_kernel<<<NHID  / (16 * MTILES_H), 256, 0, stream>>>(hidden_emb, w_kernel, 128,
                                                               h_all + (size_t)NFEAT * ADIM);

    // Stage 2: scores + segment softmax + context aggregation (one block/feature)
    attn_kernel<<<NFEAT, 128, 0, stream>>>(h_top, h_all, feat_emb, hidden_emb,
                                           w_bias, u_kernel, conn_idx, context);

    // Stage 3: out = values @ context  (split-K WMMA + deterministic reduce)
    gemm_out_kernel<<<16 * NCHUNK, 256, 0, stream>>>(values, context, part);
    reduce_kernel<<<(BATCH * DIM) / 256, 256, 0, stream>>>(part, out);
}
